// DDSPModel_79731772883060
// MI455X (gfx1250) — matrix operand order repaired and checked by reading, not compiled, not run on hardware
//
#include <hip/hip_runtime.h>
#include <hip/hip_bf16.h>

// ---------------------------------------------------------------------------
// Model constants
// ---------------------------------------------------------------------------
#define BATCH   2
#define TSTEPS  1000
#define NMFCC   30
#define HID     512
#define NHARM   64
#define NBANDS  64
#define NS      512000          // T * HOP
#define NSHALF  256000
#define NFREQ   256001          // NS/2 + 1
#define SRATE   22050.0f
#define DTF     (999.0f / 511999.0f)   // (T-1)/(NS-1)
#define INV_NS  (1.0f / 512000.0f)
#define TWO_PI  6.28318530717958647692f

typedef __attribute__((ext_vector_type(16))) _Float16 v16h;
typedef __attribute__((ext_vector_type(8)))  float    v8f;
typedef __attribute__((ext_vector_type(4)))  unsigned int v4u;
typedef __attribute__((ext_vector_type(8)))  int      v8i;
typedef __attribute__((ext_vector_type(4)))  int      v4i;

#if defined(__has_builtin)
#if __has_builtin(__builtin_amdgcn_tensor_load_to_lds) && \
    __has_builtin(__builtin_amdgcn_s_wait_tensorcnt)
#define HAVE_TDM 1
#endif
#endif
#ifndef HAVE_TDM
#define HAVE_TDM 0
#endif

// ---------------------------------------------------------------------------
// WMMA fragment helpers (CDNA5 16x16x32 f16 layouts, wave32)
// A: 16(M) x 32(K); lane&15 = M row; lanes<16 hold K {0..7,16..23},
//                                     lanes>=16 hold K {8..15,24..31}
// B: 32(K) x 16(N); lane&15 = N col; same K split as A
// Branch-free: callers guarantee 16 valid (or zero-padded) A rows.
// ---------------------------------------------------------------------------
__device__ __forceinline__ v16h frag_a(const _Float16* A, int lda, int m0,
                                       int k0, int lane) {
    int m = m0 + (lane & 15);
    int o = (lane & 16) ? 8 : 0;
    const _Float16* p = A + (size_t)m * lda + (k0 + o);
    v16h a;
#pragma unroll
    for (int j = 0; j < 8; ++j) a[j] = p[j];
#pragma unroll
    for (int j = 0; j < 8; ++j) a[8 + j] = p[16 + j];
    return a;
}

// W is row-major (N x K); B[k][n] = W[n][k]
__device__ __forceinline__ v16h frag_b(const _Float16* W, int ldw, int n0,
                                       int k0, int lane) {
    int n  = n0 + (lane & 15);
    int ko = k0 + ((lane & 16) ? 8 : 0);
    const _Float16* p = W + (size_t)n * ldw + ko;
    v16h b;
#pragma unroll
    for (int j = 0; j < 8; ++j) b[j] = p[j];
#pragma unroll
    for (int j = 0; j < 8; ++j) b[8 + j] = p[16 + j];
    return b;
}

__device__ __forceinline__ float sigm(float x) { return 1.0f / (1.0f + expf(-x)); }

#if HAVE_TDM
// ---------------------------------------------------------------------------
// TDM: DMA a 2-row x 1536-f32 tile (rows strided by TSTEPS*1536 elements)
// from global into LDS. D# per CDNA5 ISA ch.8 (group0 + group1, 2-D tile).
// ---------------------------------------------------------------------------
__device__ __forceinline__ void tdm_load_gx_rows(unsigned ldsOff, const float* gptr) {
    unsigned long long ga = (unsigned long long)(uintptr_t)gptr;
    v4u g0;
    g0[0] = 1u;                                   // count=1, user descriptor
    g0[1] = ldsOff;                               // lds byte address
    g0[2] = (unsigned)(ga & 0xffffffffu);         // global_addr[31:0]
    g0[3] = (unsigned)((ga >> 32) & 0x1ffffffu)   // global_addr[56:32]
          | (2u << 30);                           // type = 2 ("image")
    const unsigned td0    = 3 * HID;              // 1536 elements per row
    const unsigned td1    = 2;                    // two rows (m = 0,1)
    const unsigned strd0  = TSTEPS * 3 * HID;     // 1,536,000 elements
    v8i g1;
    g1[0] = (int)(2u << 16);                      // data_size = 4B, mask/flags 0
    g1[1] = (int)((td0 & 0xffffu) << 16);         // tensor_dim0[15:0]
    g1[2] = (int)((td0 >> 16) | ((td1 & 0xffffu) << 16)); // td0[31:16], td1[15:0]
    g1[3] = (int)((td1 >> 16) | ((td0 & 0xffffu) << 16)); // td1[31:16], tile_dim0
    g1[4] = (int)(td1 & 0xffffu);                 // tile_dim1=2, tile_dim2=0
    g1[5] = (int)strd0;                           // tensor_dim0_stride[31:0]
    g1[6] = 0;                                    // stride0[47:32], stride1[15:0]
    g1[7] = 0;                                    // stride1[47:16]
    v4i g2 = {0, 0, 0, 0};
    v4i g3 = {0, 0, 0, 0};
#if __clang_major__ >= 23
    v8i g4 = {0, 0, 0, 0, 0, 0, 0, 0};
    __builtin_amdgcn_tensor_load_to_lds(g0, g1, g2, g3, g4, 0);
#else
    __builtin_amdgcn_tensor_load_to_lds(g0, g1, g2, g3, 0);
#endif
}
#endif

// ---------------------------------------------------------------------------
// Generic WMMA GEMM: C[m][n] = sum_k A[m][k]*W[n][k] + bias[n]
// Requires M % 16 == 0, N % 16 == 0, K % 32 == 0 (true for all call sites).
// block = 128 threads (4 waves, 1 N-tile each), grid = (M/16, ceil(N/64))
// ---------------------------------------------------------------------------
__global__ void __launch_bounds__(128)
wmma_gemm_kernel(const _Float16* __restrict__ A, const _Float16* __restrict__ W,
                 const float* __restrict__ bias, float* __restrict__ C,
                 int M, int N, int K) {
    int lane = threadIdx.x & 31;
    int wave = threadIdx.x >> 5;
    int m0 = blockIdx.x * 16;
    int n0 = (blockIdx.y * 4 + wave) * 16;
    if (n0 >= N || m0 >= M) return;

    v8f acc = {};
    for (int k0 = 0; k0 < K; k0 += 32) {
        v16h a = frag_a(A, K, m0, k0, lane);
        v16h b = frag_b(W, K, n0, k0, lane);
        acc = __builtin_amdgcn_wmma_f32_16x16x32_f16(false, a, false, b,
                                                     (short)0, acc, false, false);
    }
    int col = n0 + (lane & 15);
    float bv = bias ? bias[col] : 0.0f;
    int rbase = m0 + ((lane & 16) ? 8 : 0);
#pragma unroll
    for (int v = 0; v < 8; ++v)
        C[(size_t)(rbase + v) * N + col] = acc[v] + bv;
}

// ---------------------------------------------------------------------------
// GRU scan: one workgroup (1024 thr = 32 waves) walks T steps.
// gh = h @ Whh^T via WMMA (zero-padded M=16, N=1536, K=512); gates via VALU.
// gx rows for step t are DMA'd into LDS by the TDM during the WMMA phase.
// ---------------------------------------------------------------------------
__global__ void __launch_bounds__(1024)
gru_scan_kernel(const float* __restrict__ gx, const _Float16* __restrict__ Whh,
                const float* __restrict__ bhh, float* __restrict__ y, int T) {
    __shared__ float    sh_h[2][HID];
    __shared__ _Float16 sh_h16[16][HID];      // rows 2..15 stay zero (WMMA pad)
    __shared__ float    sh_gh[2][3 * HID];
    __shared__ float    sh_bhh[3 * HID];
#if HAVE_TDM
    __shared__ float    sh_gx[2][3 * HID];    // TDM staging for step t
#endif

    int tid  = threadIdx.x;
    int lane = tid & 31;
    int wave = tid >> 5;

    sh_h[tid >> 9][tid & 511] = 0.0f;
    for (int i = tid; i < 16 * HID; i += 1024)
        ((_Float16*)sh_h16)[i] = (_Float16)0.0f;
    for (int i = tid; i < 3 * HID; i += 1024) sh_bhh[i] = bhh[i];
    __syncthreads();

    int m = tid >> 9;          // batch row 0/1
    int c = tid & 511;         // hidden index
    int nb = wave * 48;        // this wave's 3 N-tiles: nb, nb+16, nb+32
    const _Float16* hz = &sh_h16[0][0];
#if HAVE_TDM
    unsigned ldsGx = (unsigned)(uintptr_t)(&sh_gx[0][0]);
#endif

    for (int t = 0; t < T; ++t) {
#if HAVE_TDM
        if (wave == 0)        // one DMA per step; overlaps the WMMA phase
            tdm_load_gx_rows(ldsGx, gx + (size_t)t * (3 * HID));
#endif
        // phase 1: recurrent matmul via WMMA
        __builtin_prefetch(Whh + (size_t)nb * HID, 0, 1);
        v8f acc0 = {}, acc1 = {}, acc2 = {};
        for (int k0 = 0; k0 < HID; k0 += 32) {
            v16h a  = frag_a(hz, HID, 0, k0, lane);
            v16h b0 = frag_b(Whh, HID, nb,      k0, lane);
            acc0 = __builtin_amdgcn_wmma_f32_16x16x32_f16(false, a, false, b0,
                                                          (short)0, acc0, false, false);
            v16h b1 = frag_b(Whh, HID, nb + 16, k0, lane);
            acc1 = __builtin_amdgcn_wmma_f32_16x16x32_f16(false, a, false, b1,
                                                          (short)0, acc1, false, false);
            v16h b2 = frag_b(Whh, HID, nb + 32, k0, lane);
            acc2 = __builtin_amdgcn_wmma_f32_16x16x32_f16(false, a, false, b2,
                                                          (short)0, acc2, false, false);
        }
        if (lane < 16) {   // rows M=0 (vgpr0) and M=1 (vgpr1) live in lanes 0..15
            sh_gh[0][nb + lane]      = acc0[0];
            sh_gh[1][nb + lane]      = acc0[1];
            sh_gh[0][nb + 16 + lane] = acc1[0];
            sh_gh[1][nb + 16 + lane] = acc1[1];
            sh_gh[0][nb + 32 + lane] = acc2[0];
            sh_gh[1][nb + 32 + lane] = acc2[1];
        }
#if HAVE_TDM
        if (wave == 0)
            __builtin_amdgcn_s_wait_tensorcnt((short)0);  // gx[t] landed in LDS
#endif
        __syncthreads();

        // phase 2: gates (1024 threads == 2*512 (m,c) pairs)
#if HAVE_TDM
        const float* gxr = &sh_gx[m][0];
#else
        const float* gxr = gx + ((size_t)m * T + t) * (3 * HID);
#endif
        float r = sigm(gxr[c]           + sh_gh[m][c]           + sh_bhh[c]);
        float z = sigm(gxr[HID + c]     + sh_gh[m][HID + c]     + sh_bhh[HID + c]);
        float n = tanhf(gxr[2*HID + c]  + r * (sh_gh[m][2*HID + c] + sh_bhh[2*HID + c]));
        float hprev = sh_h[m][c];
        float hnew  = (1.0f - z) * n + z * hprev;
        __syncthreads();

        sh_h[m][c]   = hnew;
        sh_h16[m][c] = (_Float16)hnew;
        y[((size_t)m * T + t) * HID + c] = hnew;
        __syncthreads();
    }
}

// ---------------------------------------------------------------------------
// Misc small kernels
// ---------------------------------------------------------------------------
__global__ void cvt_f16_kernel(const float* __restrict__ s,
                               _Float16* __restrict__ d, int n) {
    int i = blockIdx.x * blockDim.x + threadIdx.x;
    if (i < n) d[i] = (_Float16)s[i];
}

__global__ void concat_cvt_kernel(const float* __restrict__ f0,
                                  const float* __restrict__ loud,
                                  const float* __restrict__ mfcc,
                                  _Float16* __restrict__ xin, int rows) {
    int m = blockIdx.x * blockDim.x + threadIdx.x;
    if (m >= rows) return;
    _Float16* o = xin + (size_t)m * 32;
    o[0] = (_Float16)f0[m];
    o[1] = (_Float16)loud[m];
    const float* mf = mfcc + (size_t)m * NMFCC;
    for (int i = 0; i < NMFCC; ++i) o[2 + i] = (_Float16)mf[i];
}

// LayerNorm(512) + ReLU, in place. block 256, grid = rows.
__global__ void __launch_bounds__(256)
ln_relu_kernel(float* __restrict__ x, const float* __restrict__ g,
               const float* __restrict__ beta, int rows) {
    __shared__ float red[256];
    int row = blockIdx.x, tid = threadIdx.x;
    float* xr = x + (size_t)row * HID;
    float a = xr[tid], b = xr[tid + 256];
    red[tid] = a + b;
    __syncthreads();
    for (int s = 128; s > 0; s >>= 1) {
        if (tid < s) red[tid] += red[tid + s];
        __syncthreads();
    }
    float mean = red[0] * (1.0f / HID);
    __syncthreads();
    float da = a - mean, db = b - mean;
    red[tid] = da * da + db * db;
    __syncthreads();
    for (int s = 128; s > 0; s >>= 1) {
        if (tid < s) red[tid] += red[tid + s];
        __syncthreads();
    }
    float inv = rsqrtf(red[0] * (1.0f / HID) + 1e-5f);
    float ya = da * inv * g[tid] + beta[tid];
    float yb = db * inv * g[tid + 256] + beta[tid + 256];
    xr[tid]       = ya > 0.0f ? ya : 0.0f;
    xr[tid + 256] = yb > 0.0f ? yb : 0.0f;
}

__global__ void act_harm_kernel(const float* __restrict__ h,
                                const float* __restrict__ loud,
                                float* __restrict__ amp, int n) {
    int i = blockIdx.x * blockDim.x + threadIdx.x;
    if (i >= n) return;
    int m = i >> 6;
    float x  = h[i];
    float sp = x > 20.0f ? x : log1pf(expf(x));
    amp[i] = sp * expf(loud[m] * 0.05f);
}

__global__ void act_sig_kernel(const float* __restrict__ h,
                               float* __restrict__ mag, int n) {
    int i = blockIdx.x * blockDim.x + threadIdx.x;
    if (i < n) mag[i] = sigm(h[i]);
}

// ---------------------------------------------------------------------------
// mean-interp weights (scatter-add) + mean_mag + spectral response
// ---------------------------------------------------------------------------
__global__ void interp_w_kernel(float* __restrict__ w, int ns) {
    int n = blockIdx.x * blockDim.x + threadIdx.x;
    if (n >= ns) return;
    float src = n * DTF;
    int i0 = (int)src; if (i0 > TSTEPS - 2) i0 = TSTEPS - 2;
    float fr = src - (float)i0;
    atomicAdd(&w[i0],     (1.0f - fr) * INV_NS);
    atomicAdd(&w[i0 + 1], fr * INV_NS);
}

__global__ void mean_mag_kernel(const float* __restrict__ w,
                                const float* __restrict__ mag,
                                float* __restrict__ mm) {
    int tid = threadIdx.x;
    if (tid >= BATCH * NBANDS) return;
    int b = tid >> 6, c = tid & 63;
    float acc = 0.0f;
    for (int t = 0; t < TSTEPS; ++t)
        acc += w[t] * mag[((size_t)(b * TSTEPS + t)) * NBANDS + c];
    mm[tid] = acc;
}

__global__ void resp_kernel(const float* __restrict__ mm,
                            float* __restrict__ resp, int F) {
    int f = blockIdx.x * blockDim.x + threadIdx.x;
    if (f >= F) return;
    float freq = f * (SRATE / (float)NS);
    float lg   = logf(freq + 1e-7f);
    float lf0  = logf(20.0f);
    float dln  = (logf(SRATE * 0.5f) - lf0) / 63.0f;
    float wsum = 0.0f, a0 = 0.0f, a1 = 0.0f;
    for (int c = 0; c < NBANDS; ++c) {
        float fc = expf(lf0 + c * dln);
        float d  = logf(fc + 1e-7f) - lg;
        float wv = expf(-2.0f * d * d);      // exp(-d^2 / 0.5)
        wsum += wv;
        a0 += wv * mm[c];
        a1 += wv * mm[NBANDS + c];
    }
    float inv = 1.0f / (wsum + 1e-7f);
    resp[f]     = a0 * inv;
    resp[F + f] = a1 * inv;
}

// ---------------------------------------------------------------------------
// Phase cumsum: block-level scan (1024 samples / block) + serial block offsets
// ---------------------------------------------------------------------------
__global__ void __launch_bounds__(256)
phase_partial_kernel(const float* __restrict__ f0, float* __restrict__ pp,
                     float* __restrict__ bsum) {
    __shared__ float sh[256];
    int b = blockIdx.y, tid = threadIdx.x;
    int base = blockIdx.x * 1024 + tid * 4;
    const float* f0b = f0 + (size_t)b * TSTEPS;
    float loc[4];
    float s = 0.0f;
#pragma unroll
    for (int j = 0; j < 4; ++j) {
        int n = base + j;
        float src = n * DTF;
        int i0 = (int)src; if (i0 > TSTEPS - 2) i0 = TSTEPS - 2;
        float fr = src - (float)i0;
        float v  = f0b[i0] * (1.0f - fr) + f0b[i0 + 1] * fr;
        s += v;
        loc[j] = s;
    }
    sh[tid] = s;
    __syncthreads();
    for (int off = 1; off < 256; off <<= 1) {
        float v = (tid >= off) ? sh[tid - off] : 0.0f;
        __syncthreads();
        sh[tid] += v;
        __syncthreads();
    }
    float excl = sh[tid] - s;
#pragma unroll
    for (int j = 0; j < 4; ++j)
        pp[(size_t)b * NS + base + j] = loc[j] + excl;
    if (tid == 255) bsum[b * 500 + blockIdx.x] = sh[255];
}

__global__ void scan_offsets_kernel(const float* __restrict__ bsum,
                                    float* __restrict__ boff, int nb) {
    int b = threadIdx.x;
    if (b >= BATCH) return;
    float off = 0.0f;
    for (int i = 0; i < nb; ++i) { boff[b * nb + i] = off; off += bsum[b * nb + i]; }
}

// ---------------------------------------------------------------------------
// Harmonic oscillator: 64 sins per sample (TRANS-bound)
// ---------------------------------------------------------------------------
__global__ void __launch_bounds__(256)
osc_kernel(const float* __restrict__ pp, const float* __restrict__ boff,
           const float* __restrict__ amp, float* __restrict__ harm) {
    int b = blockIdx.y;
    int n = blockIdx.x * 256 + threadIdx.x;
    float cum = pp[(size_t)b * NS + n] + boff[b * 500 + (n >> 10)];
    float ph  = cum * (TWO_PI / SRATE);
    float src = n * DTF;
    int t0 = (int)src; if (t0 > TSTEPS - 2) t0 = TSTEPS - 2;
    float fr = src - (float)t0;
    const float* A0 = amp + ((size_t)(b * TSTEPS + t0)) * NHARM;
    const float* A1 = A0 + NHARM;
    float s = 0.0f;
#pragma unroll 4
    for (int k = 0; k < NHARM; ++k) {
        float a = A0[k] + (A1[k] - A0[k]) * fr;
        s += sinf(ph * (float)(k + 1)) * a;
    }
    harm[(size_t)b * NS + n] = s;
}

// ---------------------------------------------------------------------------
// Four-step mixed-radix FFT, N = 4096 * 125.
// Step 1: 4096-pt radix-2 DIT in LDS over stride-125 columns + twiddle.
// Step 2: 125-pt direct DFT per row.
// dirSign = -1 forward, +1 inverse (1/N applied at the end).
// ---------------------------------------------------------------------------
__global__ void __launch_bounds__(1024)
fft4096_kernel(const float* __restrict__ rIn, const float2* __restrict__ cIn,
               float2* __restrict__ out, float dirSign, int useReal) {
    __shared__ float2 sh[4096];
    int b = blockIdx.y, n2 = blockIdx.x, tid = threadIdx.x;

    for (int i = tid; i < 4096; i += 1024) {
        size_t src = (size_t)b * NS + (size_t)i * 125 + n2;
        float2 v;
        if (useReal) { v = make_float2(rIn[src], 0.0f); }
        else         { v = cIn[src]; }
        int r = (int)(__brev((unsigned)i) >> 20);   // 12-bit reversal
        sh[r] = v;
    }
    __syncthreads();

    for (int s = 1; s <= 12; ++s) {
        int mlen = 1 << s, half = mlen >> 1;
        for (int i = tid; i < 2048; i += 1024) {
            int grp = i / half, pos = i - grp * half;
            int ia = grp * mlen + pos, ib = ia + half;
            float ang = dirSign * TWO_PI * (float)pos / (float)mlen;
            float wx = cosf(ang), wy = sinf(ang);
            float2 xb = sh[ib], xa = sh[ia];
            float2 tv = make_float2(wx * xb.x - wy * xb.y, wx * xb.y + wy * xb.x);
            sh[ib] = make_float2(xa.x - tv.x, xa.y - tv.y);
            sh[ia] = make_float2(xa.x + tv.x, xa.y + tv.y);
        }
        __syncthreads();
    }

    for (int k1 = tid; k1 < 4096; k1 += 1024) {
        int ph = (n2 * k1) % NS;
        float ang = dirSign * TWO_PI * (float)ph * INV_NS;
        float wx = cosf(ang), wy = sinf(ang);
        float2 v = sh[k1];
        out[(size_t)b * NS + (size_t)k1 * 125 + n2] =
            make_float2(v.x * wx - v.y * wy, v.x * wy + v.y * wx);
    }
}

__global__ void __launch_bounds__(128)
dft125_kernel(const float2* __restrict__ in, float2* __restrict__ out,
              float dirSign) {
    __shared__ float2 row[125];
    int b = blockIdx.y, r = blockIdx.x, tid = threadIdx.x;
    if (tid < 125) row[tid] = in[(size_t)b * NS + (size_t)r * 125 + tid];
    __syncthreads();
    if (tid < 125) {
        float ax = 0.0f, ay = 0.0f;
        for (int n2 = 0; n2 < 125; ++n2) {
            int ph = (n2 * tid) % 125;
            float ang = dirSign * TWO_PI * (float)ph * (1.0f / 125.0f);
            float wx = cosf(ang), wy = sinf(ang);
            float2 v = row[n2];
            ax += v.x * wx - v.y * wy;
            ay += v.x * wy + v.y * wx;
        }
        out[(size_t)b * NS + (size_t)tid * 4096 + r] = make_float2(ax, ay);
    }
}

__global__ void specmul_kernel(float2* __restrict__ spec,
                               const float* __restrict__ resp, int F) {
    int idx = blockIdx.x * blockDim.x + threadIdx.x;
    if (idx >= BATCH * NS) return;
    int b = idx / NS, k = idx - b * NS;
    int mir = (k <= NSHALF) ? k : NS - k;
    float rv = resp[(size_t)b * F + mir];
    float2 v = spec[idx];
    spec[idx] = make_float2(v.x * rv, v.y * rv);
}

__global__ void mix_kernel(const float2* __restrict__ tc,
                           const float* __restrict__ ratio,
                           float* __restrict__ outAudio,
                           const float* __restrict__ outHarm,
                           float* __restrict__ outNoise, int total) {
    int i = blockIdx.x * blockDim.x + threadIdx.x;
    if (i >= total) return;
    float nz = tc[i].x * INV_NS;
    float h  = outHarm[i];
    float rr = sigm(ratio[0]);
    outAudio[i] = rr * h + (1.0f - rr) * nz;
    outNoise[i] = nz;
}

// ---------------------------------------------------------------------------
// Host orchestration
// ---------------------------------------------------------------------------
extern "C" void kernel_launch(void* const* d_in, const int* in_sizes, int n_in,
                              void* d_out, int out_size, void* d_ws, size_t ws_size,
                              hipStream_t stream) {
    const float* f0    = (const float*)d_in[0];
    const float* loud  = (const float*)d_in[1];
    const float* mfcc  = (const float*)d_in[2];
    const float* noise = (const float*)d_in[3];
    const float* Wih0  = (const float*)d_in[4];
    const float* Whh0  = (const float*)d_in[5];
    const float* bih0  = (const float*)d_in[6];
    const float* bhh0  = (const float*)d_in[7];
    const float* Wih1  = (const float*)d_in[8];
    const float* Whh1  = (const float*)d_in[9];
    const float* bih1  = (const float*)d_in[10];
    const float* bhh1  = (const float*)d_in[11];
    const float* hW1   = (const float*)d_in[12];
    const float* hb1   = (const float*)d_in[13];
    const float* hg    = (const float*)d_in[14];
    const float* hbeta = (const float*)d_in[15];
    const float* hW2   = (const float*)d_in[16];
    const float* hb2   = (const float*)d_in[17];
    const float* nW1   = (const float*)d_in[18];
    const float* nb1   = (const float*)d_in[19];
    const float* ng    = (const float*)d_in[20];
    const float* nbeta = (const float*)d_in[21];
    const float* nW2   = (const float*)d_in[22];
    const float* nb2   = (const float*)d_in[23];
    const float* ratio = (const float*)d_in[24];

    const int M = BATCH * TSTEPS;       // 2000
    size_t off = 0;
    auto alloc = [&](size_t bytes) -> char* {
        char* p = (char*)d_ws + off;
        off += (bytes + 255) & ~(size_t)255;
        return p;
    };

    float*     gx     = (float*)alloc((size_t)M * 3 * HID * 4);
    float*     y0     = (float*)alloc((size_t)M * HID * 4);
    float*     x1     = (float*)alloc((size_t)M * HID * 4);
    float*     hbuf   = (float*)alloc((size_t)M * HID * 4);
    float*     nbuf   = (float*)alloc((size_t)M * HID * 4);
    float*     hout   = (float*)alloc((size_t)M * NHARM * 4);
    float*     nout   = (float*)alloc((size_t)M * NBANDS * 4);
    float*     hamp   = (float*)alloc((size_t)M * NHARM * 4);
    float*     fmag   = (float*)alloc((size_t)M * NBANDS * 4);
    _Float16*  xin16  = (_Float16*)alloc((size_t)M * 32 * 2);
    _Float16*  Wih0h  = (_Float16*)alloc((size_t)3 * HID * 32 * 2);
    _Float16*  Whh0h  = (_Float16*)alloc((size_t)3 * HID * HID * 2);
    _Float16*  Wih1h  = (_Float16*)alloc((size_t)3 * HID * HID * 2);
    _Float16*  Whh1h  = (_Float16*)alloc((size_t)3 * HID * HID * 2);
    _Float16*  hW1h   = (_Float16*)alloc((size_t)HID * HID * 2);
    _Float16*  hW2h   = (_Float16*)alloc((size_t)NHARM * HID * 2);
    _Float16*  nW1h   = (_Float16*)alloc((size_t)HID * HID * 2);
    _Float16*  nW2h   = (_Float16*)alloc((size_t)NBANDS * HID * 2);
    _Float16*  y0h    = (_Float16*)alloc((size_t)M * HID * 2);
    _Float16*  x1h    = (_Float16*)alloc((size_t)M * HID * 2);
    _Float16*  hh16   = (_Float16*)alloc((size_t)M * HID * 2);
    _Float16*  nh16   = (_Float16*)alloc((size_t)M * HID * 2);
    float*     wbuf   = (float*)alloc((size_t)TSTEPS * 4);
    float*     mmbuf  = (float*)alloc((size_t)BATCH * NBANDS * 4);
    float*     respb  = (float*)alloc((size_t)BATCH * NFREQ * 4);
    float*     pp     = (float*)alloc((size_t)BATCH * NS * 4);
    float*     bsum   = (float*)alloc((size_t)BATCH * 500 * 4);
    float*     boff   = (float*)alloc((size_t)BATCH * 500 * 4);
    float2*    specA  = (float2*)alloc((size_t)BATCH * NS * 8);
    float2*    specB  = (float2*)alloc((size_t)BATCH * NS * 8);

    float* outAudio = (float*)d_out;
    float* outHarm  = outAudio + (size_t)BATCH * NS;
    float* outNoise = outAudio + (size_t)2 * BATCH * NS;

    auto cvt = [&](const float* s, _Float16* d, int n) {
        cvt_f16_kernel<<<(n + 255) / 256, 256, 0, stream>>>(s, d, n);
    };

    // ---- weight + activation conversions
    cvt(Wih0, Wih0h, 3 * HID * 32);
    cvt(Whh0, Whh0h, 3 * HID * HID);
    cvt(Wih1, Wih1h, 3 * HID * HID);
    cvt(Whh1, Whh1h, 3 * HID * HID);
    cvt(hW1, hW1h, HID * HID);
    cvt(hW2, hW2h, NHARM * HID);
    cvt(nW1, nW1h, HID * HID);
    cvt(nW2, nW2h, NBANDS * HID);
    concat_cvt_kernel<<<(M + 255) / 256, 256, 0, stream>>>(f0, loud, mfcc, xin16, M);

    // ---- GRU layer 0
    wmma_gemm_kernel<<<dim3(M / 16, (3 * HID) / 64), 128, 0, stream>>>(
        xin16, Wih0h, bih0, gx, M, 3 * HID, 32);
    gru_scan_kernel<<<1, 1024, 0, stream>>>(gx, Whh0h, bhh0, y0, TSTEPS);

    // ---- GRU layer 1
    cvt(y0, y0h, M * HID);
    wmma_gemm_kernel<<<dim3(M / 16, (3 * HID) / 64), 128, 0, stream>>>(
        y0h, Wih1h, bih1, gx, M, 3 * HID, HID);
    gru_scan_kernel<<<1, 1024, 0, stream>>>(gx, Whh1h, bhh1, x1, TSTEPS);
    cvt(x1, x1h, M * HID);

    // ---- harmonic head
    wmma_gemm_kernel<<<dim3(M / 16, HID / 64), 128, 0, stream>>>(
        x1h, hW1h, hb1, hbuf, M, HID, HID);
    ln_relu_kernel<<<M, 256, 0, stream>>>(hbuf, hg, hbeta, M);
    cvt(hbuf, hh16, M * HID);
    wmma_gemm_kernel<<<dim3(M / 16, 1), 128, 0, stream>>>(
        hh16, hW2h, hb2, hout, M, NHARM, HID);
    act_harm_kernel<<<(M * NHARM + 255) / 256, 256, 0, stream>>>(
        hout, loud, hamp, M * NHARM);

    // ---- noise head
    wmma_gemm_kernel<<<dim3(M / 16, HID / 64), 128, 0, stream>>>(
        x1h, nW1h, nb1, nbuf, M, HID, HID);
    ln_relu_kernel<<<M, 256, 0, stream>>>(nbuf, ng, nbeta, M);
    cvt(nbuf, nh16, M * HID);
    wmma_gemm_kernel<<<dim3(M / 16, 1), 128, 0, stream>>>(
        nh16, nW2h, nb2, nout, M, NBANDS, HID);
    act_sig_kernel<<<(M * NBANDS + 255) / 256, 256, 0, stream>>>(
        nout, fmag, M * NBANDS);

    // ---- filtered-noise response
    hipMemsetAsync(wbuf, 0, (size_t)TSTEPS * 4, stream);
    interp_w_kernel<<<(NS + 255) / 256, 256, 0, stream>>>(wbuf, NS);
    mean_mag_kernel<<<1, 128, 0, stream>>>(wbuf, fmag, mmbuf);
    resp_kernel<<<(NFREQ + 255) / 256, 256, 0, stream>>>(mmbuf, respb, NFREQ);

    // ---- harmonic oscillator
    phase_partial_kernel<<<dim3(500, BATCH), 256, 0, stream>>>(f0, pp, bsum);
    scan_offsets_kernel<<<1, 32, 0, stream>>>(bsum, boff, 500);
    osc_kernel<<<dim3(NS / 256, BATCH), 256, 0, stream>>>(pp, boff, hamp, outHarm);

    // ---- FFT round trip (forward, filter, inverse)
    fft4096_kernel<<<dim3(125, BATCH), 1024, 0, stream>>>(
        noise, (const float2*)nullptr, specA, -1.0f, 1);
    dft125_kernel<<<dim3(4096, BATCH), 128, 0, stream>>>(specA, specB, -1.0f);
    specmul_kernel<<<(BATCH * NS + 255) / 256, 256, 0, stream>>>(specB, respb, NFREQ);
    fft4096_kernel<<<dim3(125, BATCH), 1024, 0, stream>>>(
        (const float*)nullptr, specB, specA, 1.0f, 0);
    dft125_kernel<<<dim3(4096, BATCH), 128, 0, stream>>>(specA, specB, 1.0f);

    // ---- mix + outputs
    mix_kernel<<<(BATCH * NS + 255) / 256, 256, 0, stream>>>(
        specB, ratio, outAudio, outHarm, outNoise, BATCH * NS);
}
